// CountNCELoss_56289841381701
// MI455X (gfx1250) — matrix operand order repaired and checked by reading, not compiled, not run on hardware
//
#include <hip/hip_runtime.h>
#include <hip/hip_bf16.h>

#define B_   32
#define L_   1024
#define CD   256
#define ZD   256
#define P_   12
#define NEG_ 8
#define NSH  9   // 1 positive + 8 negative shifts
#define MT   32  // rows of l per block (two 16-row WMMA subtiles)

typedef __attribute__((ext_vector_type(16))) __bf16 v16bf;
typedef __attribute__((ext_vector_type(8)))  float  v8f;

union V16U { int4 q[2]; v16bf v; };

__device__ __forceinline__ unsigned short f2bf_bits(float f) {
    union { float f; unsigned u; } x; x.f = f;
    unsigned r = x.u + 0x7FFFu + ((x.u >> 16) & 1u);  // round-to-nearest-even
    return (unsigned short)(r >> 16);
}

// ---- prep: 1/sum(length) and zero the output accumulator --------------------
__global__ void prep_scalar(const int* __restrict__ length, float* __restrict__ meta,
                            float* __restrict__ out) {
    int s = 0;
    for (int i = 0; i < B_; ++i) s += length[i];
    meta[0] = 1.0f / (float)s;
    out[0]  = 0.0f;
}

// ---- prep: f32 -> bf16 bit conversion ---------------------------------------
__global__ void cvt_bf16(const float* __restrict__ src, unsigned short* __restrict__ dst, int n) {
    int i = blockIdx.x * blockDim.x + threadIdx.x;
    int stride = gridDim.x * blockDim.x;
    for (; i < n; i += stride) dst[i] = f2bf_bits(src[i]);
}

// ---- prep: W[p][c][z] -> WT[p][z][c] in bf16 (B-operand wants K=c contiguous)
__global__ void cvt_transpose_w(const float* __restrict__ W, unsigned short* __restrict__ WT) {
    int i = blockIdx.x * blockDim.x + threadIdx.x;   // indexes output, coalesced writes
    if (i >= P_ * CD * ZD) return;
    int k = i & (CD - 1);          // c index (K)
    int n = (i >> 8) & (ZD - 1);   // z index (N / column)
    int p = i >> 16;
    WT[i] = f2bf_bits(W[(p * CD + k) * ZD + n]);
}

// ---- main fused kernel: one wave handles (p, b, 32-row l-tile) --------------
__global__ __launch_bounds__(32) void nce_main(
    const unsigned short* __restrict__ cB,   // [B,L,CD] bf16
    const unsigned short* __restrict__ zB,   // [B,L,ZD] bf16
    const unsigned short* __restrict__ WT,   // [P,ZD,CD] bf16 (transposed)
    const int*            __restrict__ neg_shift,
    const int*            __restrict__ length,
    const float*          __restrict__ meta,
    float*                __restrict__ out)
{
    __shared__ alignas(16) unsigned short WcL[MT * CD];   // 32x256 bf16 = 16 KB

    const int blk = blockIdx.x;
    const int lt  = blk & 31;               // L_/MT = 32 tiles
    const int b   = (blk >> 5) & 31;
    const int p   = blk >> 10;              // 0..11
    const int l0  = lt * MT;

    const int lane = threadIdx.x & 31;
    const int ln   = lane & 15;
    const int hi   = lane >> 4;

    const float inv  = meta[0];
    const int   lenb = length[b];

    int sh[NSH];
    sh[0] = p + 1;
#pragma unroll
    for (int i = 0; i < NEG_; ++i) sh[i + 1] = neg_shift[i];

    // ---------------- Phase 1: Wc = c_tile[32xC] @ W[p][CxZ] -> LDS (bf16) ---
    // A layout (16-bit 16x32): lane row = ln; halfs [K0..7 | K16..23] (lo lanes)
    // or [K8..15 | K24..31] (hi lanes). Two 16-row A tiles held in registers.
    v16bf Ac0[8], Ac1[8];
    {
        const unsigned short* ar0 = cB + (size_t)(b * L_ + l0 + ln) * CD;
        const unsigned short* ar1 = ar0 + 16 * CD;
#pragma unroll
        for (int kc = 0; kc < 8; ++kc) {
            const int ka0 = kc * 32 + (hi ? 8 : 0);
            V16U t0, t1;
            t0.q[0] = *(const int4*)(ar0 + ka0);
            t0.q[1] = *(const int4*)(ar0 + ka0 + 16);
            t1.q[0] = *(const int4*)(ar1 + ka0);
            t1.q[1] = *(const int4*)(ar1 + ka0 + 16);
            Ac0[kc] = t0.v;
            Ac1[kc] = t1.v;
        }
    }

    const unsigned short* wbase = WT + (size_t)p * ZD * CD;
    for (int nt = 0; nt < 16; ++nt) {
        v8f acc0 = {0.f, 0.f, 0.f, 0.f, 0.f, 0.f, 0.f, 0.f};
        v8f acc1 = {0.f, 0.f, 0.f, 0.f, 0.f, 0.f, 0.f, 0.f};
        // B layout (32x16): lane column = ln; K halfs [K0..7 | K16..23] (lo) or [K8..15 | K24..31] (hi)
        const unsigned short* bcol = wbase + (size_t)(nt * 16 + ln) * CD + (hi ? 8 : 0);
#pragma unroll
        for (int kc = 0; kc < 8; ++kc) {
            V16U t;
            t.q[0] = *(const int4*)(bcol + kc * 32);
            t.q[1] = *(const int4*)(bcol + kc * 32 + 16);
            acc0 = __builtin_amdgcn_wmma_f32_16x16x32_bf16(
                       false, Ac0[kc], false, t.v, (short)0, acc0, false, false);
            acc1 = __builtin_amdgcn_wmma_f32_16x16x32_bf16(
                       false, Ac1[kc], false, t.v, (short)0, acc1, false, false);
        }
        // D layout: VGPR v, lanes0-15 -> (M=v, N=ln); lanes16-31 -> (M=v+8, N=ln)
#pragma unroll
        for (int v = 0; v < 8; ++v) {
            const int m = v + 8 * hi;
            WcL[m * CD + nt * 16 + ln]        = f2bf_bits(acc0[v]);
            WcL[(16 + m) * CD + nt * 16 + ln] = f2bf_bits(acc1[v]);
        }
    }
    __syncthreads();

    // ------------- Phase 2+3: per 16-row subtile, scores + logsumexp ---------
    const size_t zbase = (size_t)b * L_ * ZD;
    // Which diagonal element (if any) this lane owns: D(M,N) at (VGPR v, lane)
    // has M = v + 8*hi, N = ln; diagonal needs ln == v + 8*hi  =>  v = ln - 8*hi.
    const int  vsel  = ln - 8 * hi;                 // valid when 0 <= vsel < 8
    const bool owner = (vsel >= 0) && (vsel < 8);   // 16 owner lanes, row m = ln

    float vtot = 0.0f;
#pragma unroll 1
    for (int t = 0; t < 2; ++t) {
        v8f sacc[NSH];
#pragma unroll
        for (int j = 0; j < NSH; ++j)
            sacc[j] = (v8f){0.f, 0.f, 0.f, 0.f, 0.f, 0.f, 0.f, 0.f};

        const int lsub = l0 + t * 16;
        for (int kc = 0; kc < 8; ++kc) {
            const int ka0 = kc * 32 + (hi ? 8 : 0);
            v16bf aw;
            {
                V16U ta;
                ta.q[0] = *(const int4*)(WcL + (t * 16 + ln) * CD + ka0);
                ta.q[1] = *(const int4*)(WcL + (t * 16 + ln) * CD + ka0 + 16);
                aw = ta.v;
            }
            const int kb0 = kc * 32 + (hi ? 8 : 0);
#pragma unroll
            for (int j = 0; j < NSH; ++j) {
                const int zr = (lsub + sh[j] + ln) & (L_ - 1);   // jnp.roll wraps
                const unsigned short* zp = zB + zbase + (size_t)zr * ZD + kb0;
                V16U t2;
                t2.q[0] = *(const int4*)(zp);
                t2.q[1] = *(const int4*)(zp + 16);
                sacc[j] = __builtin_amdgcn_wmma_f32_16x16x32_bf16(
                              false, aw, false, t2.v, (short)0, sacc[j], false, false);
            }
        }

        // Extract diagonal in-register (cndmask tree), logsumexp on owner lanes
        float s[NSH];
#pragma unroll
        for (int j = 0; j < NSH; ++j) {
            float d = 0.0f;
#pragma unroll
            for (int v = 0; v < 8; ++v) d = (vsel == v) ? sacc[j][v] : d;
            s[j] = d;
        }
        float mx = -1e30f;
#pragma unroll
        for (int j = 0; j < NSH; ++j) mx = fmaxf(mx, s[j]);
        float sum = 0.0f;
#pragma unroll
        for (int j = 0; j < NSH; ++j) sum += __expf(s[j] - mx);
        const float lse  = mx + __logf(sum);
        const float loss = lse - s[0];                 // s[0] is the positive
        const int   lrow = lsub + ln;                  // owner lane's row is ln
        if (owner && (lrow < lenb)) vtot += loss * inv;
    }

    // wave reduction + single atomic per block
#pragma unroll
    for (int o = 16; o > 0; o >>= 1) vtot += __shfl_down(vtot, o, 32);
    if (lane == 0) atomicAdd(out, vtot);
}

extern "C" void kernel_launch(void* const* d_in, const int* in_sizes, int n_in,
                              void* d_out, int out_size, void* d_ws, size_t ws_size,
                              hipStream_t stream) {
    const float* c         = (const float*)d_in[0];
    const float* z         = (const float*)d_in[1];
    const float* W         = (const float*)d_in[2];
    const int*   neg_shift = (const int*)d_in[3];
    const int*   length    = (const int*)d_in[4];
    float*       out       = (float*)d_out;

    unsigned char* ws = (unsigned char*)d_ws;
    unsigned short* cB   = (unsigned short*)(ws);                 // 16 MB
    unsigned short* zB   = (unsigned short*)(ws + 16777216);      // 16 MB
    unsigned short* WT   = (unsigned short*)(ws + 33554432);      // 1.5 MB
    float*          meta = (float*)(ws + 35127296);

    prep_scalar<<<1, 1, 0, stream>>>(length, meta, out);

    const int nc = B_ * L_ * CD;
    cvt_bf16<<<(nc + 255) / 256, 256, 0, stream>>>(c, cB, nc);
    cvt_bf16<<<(nc + 255) / 256, 256, 0, stream>>>(z, zB, nc);

    const int nw = P_ * CD * ZD;
    cvt_transpose_w<<<(nw + 255) / 256, 256, 0, stream>>>(W, WT);

    nce_main<<<P_ * B_ * (L_ / MT), 32, 0, stream>>>(cB, zB, WT, neg_shift, length, meta, out);
}
